// Head_43336220017098
// MI455X (gfx1250) — matrix operand order repaired and checked by reading, not compiled, not run on hardware
//
#include <hip/hip_runtime.h>
#include <hip/hip_bf16.h>

// ---------------------------------------------------------------------------
// Causal attention head for MI455X (gfx1250, wave32, WMMA).
//   B=8, T=2048, C=1024, H=64.
// Stage 1: qkv projection GEMM  [B*T, C] x [C, 3H] in bf16 WMMA, fp32 acc.
// Stage 2: flash attention (online softmax), bf16 WMMA, fp32 acc/softmax.
// ---------------------------------------------------------------------------

typedef __attribute__((ext_vector_type(16))) __bf16 bf16x16;
typedef __attribute__((ext_vector_type(8)))  float  f32x8;

union Frag {
    unsigned u[8];
    uint4    q[2];
    bf16x16  v;
};

__device__ __forceinline__ unsigned short f2bf(float a) {
    union { __bf16 h; unsigned short u; } r;
    r.h = (__bf16)a;
    return r.u;
}
__device__ __forceinline__ unsigned pack_bf16(float a, float b) {
    union { __bf16 h[2]; unsigned u; } r;
    r.h[0] = (__bf16)a;
    r.h[1] = (__bf16)b;
    return r.u;
}

// ds_swizzle xor-reduction inside each 16-lane half (and=0x1f, or=0, xor=m).
#define SWZF(v, imm)                                                          \
    __builtin_bit_cast(float, __builtin_amdgcn_ds_swizzle(                    \
                                  __builtin_bit_cast(int, (v)), (imm)))

#define NB 8
#define NT 2048
#define NC 1024
#define NH 64

// ---------------------------------------------------------------------------
// Kernel 1: q/k/v projections.
// Grid: 128 blocks x 256 threads (8 waves). Each wave: 16 rows, 12 N-tiles.
// Outputs (bf16): k_ws,q_ws as [B*T][64]; v transposed as vt_ws [B*64][T].
// ---------------------------------------------------------------------------
__global__ __launch_bounds__(256) void qkv_proj_kernel(
    const float* __restrict__ x,
    const float* __restrict__ Wk, const float* __restrict__ Wq,
    const float* __restrict__ Wv,
    unsigned short* __restrict__ k_ws, unsigned short* __restrict__ q_ws,
    unsigned short* __restrict__ vt_ws)
{
    // W chunk staged as bf16: [j=0..191][c=0..31]; c contiguous so
    // B-fragments are two ds_load_b128 per tile.
    __shared__ __align__(16) unsigned short lds_w[192 * 32];

    const int tid  = threadIdx.x;
    const int wave = tid >> 5;
    const int lane = tid & 31;
    const int half = lane >> 4;   // which 16-lane half of the wave
    const int l16  = lane & 15;
    const int rowbase = blockIdx.x * 128 + wave * 16;
    const int arow    = rowbase + l16;          // A-matrix row for this lane

    f32x8 acc[12];
#pragma unroll
    for (int i = 0; i < 12; ++i)
#pragma unroll
        for (int e = 0; e < 8; ++e) acc[i][e] = 0.0f;

    for (int kb = 0; kb < NC; kb += 32) {
        __syncthreads();
        // Stage 32x192 weight chunk -> LDS bf16.
        // 1536 quad-groups (4 consecutive c each); p fixed per unrolled it.
#pragma unroll
        for (int it = 0; it < 6; ++it) {
            const int p  = it >> 1;                       // compile-time
            const int gl = tid + (it & 1) * 256;          // 0..511
            const int h  = gl >> 3;                       // 0..63
            const int c0 = (gl & 7) * 4;                  // 0,4,..,28
            const float* wp = (p == 0) ? Wk : ((p == 1) ? Wq : Wv);
            const float f0 = wp[(size_t)(kb + c0 + 0) * NH + h];
            const float f1 = wp[(size_t)(kb + c0 + 1) * NH + h];
            const float f2 = wp[(size_t)(kb + c0 + 2) * NH + h];
            const float f3 = wp[(size_t)(kb + c0 + 3) * NH + h];
            uint2 d;
            d.x = pack_bf16(f0, f1);
            d.y = pack_bf16(f2, f3);
            *(uint2*)&lds_w[(p * 64 + h) * 32 + c0] = d;
        }
        __syncthreads();

        // A fragment (16x32 bf16): lane<16 holds K {0..7,16..23}, lane>=16
        // holds K {8..15,24..31}; b128 loads + native pack.
        Frag a;
        {
            const float* px = x + (size_t)arow * NC + kb + (half ? 8 : 0);
            const float4 fa = *(const float4*)(px + 0);
            const float4 fb = *(const float4*)(px + 4);
            const float4 fc = *(const float4*)(px + 16);
            const float4 fd = *(const float4*)(px + 20);
            a.u[0] = pack_bf16(fa.x, fa.y);
            a.u[1] = pack_bf16(fa.z, fa.w);
            a.u[2] = pack_bf16(fb.x, fb.y);
            a.u[3] = pack_bf16(fb.z, fb.w);
            a.u[4] = pack_bf16(fc.x, fc.y);
            a.u[5] = pack_bf16(fc.z, fc.w);
            a.u[6] = pack_bf16(fd.x, fd.y);
            a.u[7] = pack_bf16(fd.z, fd.w);
        }

        // 2-deep ping-pong: issue loads for tile nt+1 before the WMMA for
        // tile nt, so each WMMA waits only on loads issued one step earlier.
        const int bbase = l16 * 32 + (half ? 8 : 0);
        Frag bcur, bnxt;
        bcur.q[0] = *(const uint4*)&lds_w[bbase];
        bcur.q[1] = *(const uint4*)&lds_w[bbase + 16];
#pragma unroll
        for (int nt = 0; nt < 12; ++nt) {
            if (nt + 1 < 12) {
                const int boff = bbase + (nt + 1) * 16 * 32;
                bnxt.q[0] = *(const uint4*)&lds_w[boff];
                bnxt.q[1] = *(const uint4*)&lds_w[boff + 16];
            }
            acc[nt] = __builtin_amdgcn_wmma_f32_16x16x32_bf16(
                false, a.v, false, bcur.v, (short)0, acc[nt], false, false);
            bcur = bnxt;
        }
    }

    // Epilogue: C layout -> lane holds col n=l16, rows m=r+8*half.
#pragma unroll
    for (int nt = 0; nt < 12; ++nt) {
        const int p = nt >> 2;
        const int h = (nt & 3) * 16 + l16;
#pragma unroll
        for (int r = 0; r < 8; ++r) {
            const int m  = r + (half ? 8 : 0);
            const int gr = rowbase + m;
            const unsigned short val = f2bf(acc[nt][r]);
            if (p == 0) {
                k_ws[(size_t)gr * NH + h] = val;
            } else if (p == 1) {
                q_ws[(size_t)gr * NH + h] = val;
            } else {
                const int b = gr >> 11, t = gr & (NT - 1);
                vt_ws[((size_t)b * NH + h) * NT + t] = val;
            }
        }
    }
}

// ---------------------------------------------------------------------------
// Kernel 2: causal flash attention, one wave per 16-query tile.
// Grid: B*T/16 = 1024 blocks x 32 threads.
// ---------------------------------------------------------------------------
__global__ __launch_bounds__(32) void flash_attn_kernel(
    const unsigned short* __restrict__ k_ws,
    const unsigned short* __restrict__ q_ws,
    const unsigned short* __restrict__ vt_ws,
    float* __restrict__ out)
{
    // P tile bounce buffer for C-layout -> A-layout transpose: 16 x 32 bf16.
    __shared__ __align__(16) unsigned short lds_p[16 * 32];

    const int lane = threadIdx.x & 31;
    const int half = lane >> 4;
    const int l16  = lane & 15;
    const int wg   = blockIdx.x;
    const int b    = wg >> 7;            // 128 query tiles per batch
    const int t0   = (wg & 127) << 4;

    // Q as two 16x32 A fragments (h 0..31 and 32..63), loaded once.
    Frag qf[2];
    {
        const unsigned short* qrow =
            q_ws + ((size_t)b * NT + (t0 + l16)) * NH;
#pragma unroll
        for (int f = 0; f < 2; ++f) {
            const int h0 = f * 32 + (half ? 8 : 0);
            qf[f].q[0] = *(const uint4*)&qrow[h0];
            qf[f].q[1] = *(const uint4*)&qrow[h0 + 16];
        }
    }

    f32x8 o[4];
#pragma unroll
    for (int i = 0; i < 4; ++i)
#pragma unroll
        for (int e = 0; e < 8; ++e) o[i][e] = 0.0f;

    float mrow[8], lrow[8];
#pragma unroll
    for (int r = 0; r < 8; ++r) { mrow[r] = -1e30f; lrow[r] = 0.0f; }

    const float scale = 0.03125f;  // 1/sqrt(C) = 1/32

    for (int st = 0; st <= t0 + 15; st += 32) {
        // ---- S = Q * K^T (two 16x16 C tiles), ping-ponged K fragments ----
        const unsigned short* kbase = k_ws + (size_t)b * NT * NH;
        f32x8 s[2];
        Frag kcur, knxt;
        {
            int key  = st + l16;
            int keyc = (key > NT - 1) ? (NT - 1) : key;
            const int h0 = (half ? 8 : 0);
            kcur.q[0] = *(const uint4*)&kbase[(size_t)keyc * NH + h0];
            kcur.q[1] = *(const uint4*)&kbase[(size_t)keyc * NH + h0 + 16];
        }
#pragma unroll
        for (int i = 0; i < 4; ++i) {       // i = tile*2 + f
            const int tile = i >> 1, f = i & 1;
            if (i + 1 < 4) {
                const int ntile = (i + 1) >> 1, nf = (i + 1) & 1;
                int key  = st + ntile * 16 + l16;
                int keyc = (key > NT - 1) ? (NT - 1) : key;
                const int h0 = nf * 32 + (half ? 8 : 0);
                knxt.q[0] = *(const uint4*)&kbase[(size_t)keyc * NH + h0];
                knxt.q[1] = *(const uint4*)&kbase[(size_t)keyc * NH + h0 + 16];
            }
            if (f == 0) {
#pragma unroll
                for (int e = 0; e < 8; ++e) s[tile][e] = 0.0f;
            }
            s[tile] = __builtin_amdgcn_wmma_f32_16x16x32_bf16(
                false, qf[f].v, false, kcur.v, (short)0, s[tile],
                false, false);
            kcur = knxt;
        }

        // ---- issue V-fragment loads NOW: they overlap the softmax ----
        int s0 = st + (half ? 16 : 0);
        if (s0 > NT - 16) s0 = NT - 16;   // clamp, keeps 16-alignment
        Frag vb[4];
#pragma unroll
        for (int nt = 0; nt < 4; ++nt) {
            const unsigned short* vrow =
                vt_ws + ((size_t)b * NH + (nt * 16 + l16)) * NT + s0;
            vb[nt].q[0] = *(const uint4*)&vrow[0];
            vb[nt].q[1] = *(const uint4*)&vrow[8];
        }

        // ---- scale + causal mask ----
        const bool needmask = (st + 31) >= t0;
#pragma unroll
        for (int tile = 0; tile < 2; ++tile) {
            const int key = st + tile * 16 + l16;
#pragma unroll
            for (int r = 0; r < 8; ++r) {
                const int m = r + (half ? 8 : 0);
                float sv = s[tile][r] * scale;
                if (needmask && (key > t0 + m)) sv = -1e30f;
                s[tile][r] = sv;
            }
        }

        // ---- online softmax (row stats live per 16-lane half) ----
        float alpha[8];
#pragma unroll
        for (int r = 0; r < 8; ++r) {
            float mx = fmaxf(s[0][r], s[1][r]);
            mx = fmaxf(mx, SWZF(mx, (1 << 10) | 0x1f));
            mx = fmaxf(mx, SWZF(mx, (2 << 10) | 0x1f));
            mx = fmaxf(mx, SWZF(mx, (4 << 10) | 0x1f));
            mx = fmaxf(mx, SWZF(mx, (8 << 10) | 0x1f));
            const float mn = fmaxf(mrow[r], mx);
            alpha[r] = __expf(mrow[r] - mn);
            const float p0 = __expf(s[0][r] - mn);
            const float p1 = __expf(s[1][r] - mn);
            s[0][r] = p0; s[1][r] = p1;
            float sum = p0 + p1;
            sum += SWZF(sum, (1 << 10) | 0x1f);
            sum += SWZF(sum, (2 << 10) | 0x1f);
            sum += SWZF(sum, (4 << 10) | 0x1f);
            sum += SWZF(sum, (8 << 10) | 0x1f);
            lrow[r] = lrow[r] * alpha[r] + sum;
            mrow[r] = mn;
        }

        // ---- rescale accumulator ----
#pragma unroll
        for (int nt = 0; nt < 4; ++nt)
#pragma unroll
            for (int r = 0; r < 8; ++r) o[nt][r] *= alpha[r];

        // ---- P: C layout -> LDS -> A-fragment layout ----
#pragma unroll
        for (int r = 0; r < 8; ++r) {
            const int m = r + (half ? 8 : 0);
            lds_p[m * 32 + l16]      = f2bf(s[0][r]);
            lds_p[m * 32 + 16 + l16] = f2bf(s[1][r]);
        }
        __syncthreads();
        Frag pf;
        {
            const int c0 = l16 * 32 + (half ? 16 : 0);
            pf.q[0] = *(const uint4*)&lds_p[c0];
            pf.q[1] = *(const uint4*)&lds_p[c0 + 8];
        }
        __syncthreads();

        // ---- O += P * V ----
#pragma unroll
        for (int nt = 0; nt < 4; ++nt) {
            o[nt] = __builtin_amdgcn_wmma_f32_16x16x32_bf16(
                false, pf.v, false, vb[nt].v, (short)0, o[nt], false, false);
        }
    }

    // ---- normalize + store fp32 output ----
#pragma unroll
    for (int nt = 0; nt < 4; ++nt) {
        const int h = nt * 16 + l16;
#pragma unroll
        for (int r = 0; r < 8; ++r) {
            const int m = r + (half ? 8 : 0);
            out[((size_t)b * NT + t0 + m) * NH + h] = o[nt][r] / lrow[r];
        }
    }
}

// ---------------------------------------------------------------------------
extern "C" void kernel_launch(void* const* d_in, const int* in_sizes, int n_in,
                              void* d_out, int out_size, void* d_ws,
                              size_t ws_size, hipStream_t stream)
{
    (void)in_sizes; (void)n_in; (void)out_size; (void)ws_size;
    const float* x  = (const float*)d_in[0];
    const float* Wk = (const float*)d_in[1];
    const float* Wq = (const float*)d_in[2];
    const float* Wv = (const float*)d_in[3];

    // Workspace layout (bf16): k [B*T*64], q [B*T*64], v^T [B*64*T] = 6 MB.
    unsigned short* k_ws  = (unsigned short*)d_ws;
    unsigned short* q_ws  = k_ws + (size_t)NB * NT * NH;
    unsigned short* vt_ws = q_ws + (size_t)NB * NT * NH;

    qkv_proj_kernel<<<dim3((NB * NT) / 128), dim3(256), 0, stream>>>(
        x, Wk, Wq, Wv, k_ws, q_ws, vt_ws);
    flash_attn_kernel<<<dim3((NB * NT) / 16), dim3(32), 0, stream>>>(
        k_ws, q_ws, vt_ws, (float*)d_out);
}
